// GINConv_39247411151300
// MI455X (gfx1250) — compile-verified, matrix operand order checked by reading, not run on hardware
//
#include <hip/hip_runtime.h>
#include <hip/hip_bf16.h>

typedef __attribute__((ext_vector_type(16))) _Float16 v16h;
typedef __attribute__((ext_vector_type(8)))  _Float16 v8h;
typedef __attribute__((ext_vector_type(8)))  float    v8f;

#define D_FEAT 128
#define ROWS_PER_BLOCK 64   // 4 M-tiles of 16 per block

// ---------------------------------------------------------------------------
// Kernel 1: h0 = x   (vectorized float4 copy into workspace)
// ---------------------------------------------------------------------------
__global__ void init_h0(const float4* __restrict__ x, float4* __restrict__ h0, int n4) {
    int i = blockIdx.x * blockDim.x + threadIdx.x;
    if (i < n4) h0[i] = x[i];
}

// ---------------------------------------------------------------------------
// Kernel 2: scatter-add  h0[row] += x[col]  for row != col
// One wave (32 lanes) per edge (e is wave-uniform -> scalarized index loads),
// 4 features per lane -> 4 no-return global_atomic_add_f32 per lane.
// ---------------------------------------------------------------------------
__global__ void scatter_edges(const float* __restrict__ x,
                              const long long* __restrict__ ei,
                              float* __restrict__ h0, int E) {
    int t = blockIdx.x * blockDim.x + threadIdx.x;
    int e = t >> 5;
    if (e >= E) return;
    int lane = t & 31;
    int r = (int)ei[e];
    int c = (int)ei[(long long)E + e];
    if (r == c) return;
    const float4 v = *(const float4*)(x + c * D_FEAT + lane * 4);
    float* dst = h0 + r * D_FEAT + lane * 4;
    unsafeAtomicAdd(dst + 0, v.x);
    unsafeAtomicAdd(dst + 1, v.y);
    unsafeAtomicAdd(dst + 2, v.z);
    unsafeAtomicAdd(dst + 3, v.w);
}

// ---------------------------------------------------------------------------
// Kernel 3: pre-swizzle a 128x128 f32 weight (row-major W[k][n]) into f16
// WMMA B-fragments (ISA 7.12.2 16-bit layout, wave32):
//   lane = (n & 15) + 16*khalf ; elems 0..7 = K kbase+0..7, 8..15 = K kbase+16..23
// Fragment (nt, kt) stored as 32 contiguous bytes per lane.
// ---------------------------------------------------------------------------
__global__ void prep_w(const float* __restrict__ W, _Float16* __restrict__ wf) {
    int t = blockIdx.x * blockDim.x + threadIdx.x;   // 0..1023
    if (t >= 8 * 4 * 32) return;
    int lane  = t & 31;
    int kt    = (t >> 5) & 3;
    int nt    = t >> 7;
    int col   = nt * 16 + (lane & 15);
    int kbase = kt * 32 + ((lane >> 4) << 3);
    _Float16* dst = wf + (((nt * 4 + kt) * 32 + lane) << 4);
#pragma unroll
    for (int e = 0; e < 8; ++e) {
        dst[e]     = (_Float16)W[(kbase + e)      * D_FEAT + col];
        dst[8 + e] = (_Float16)W[(kbase + 16 + e) * D_FEAT + col];
    }
}

// ---------------------------------------------------------------------------
// GEMM: C[64 x 128 tile] = act(A[64x128] @ Wfrag + bias)
// 256 threads = 8 waves; wave w owns N-tile w (16 cols) across 4 M-tiles.
// Per K-step: 1 B-fragment load, then ALL 4 A-fragment LDS reads (clause, one
// wait), then 4 back-to-back WMMAs reusing B (no WMMA->WMMA hazard: distinct D).
// ---------------------------------------------------------------------------
template <typename SrcT, bool RELU, bool OUT_F16>
__global__ __launch_bounds__(256)
void gemm_gin(const SrcT* __restrict__ A, const _Float16* __restrict__ wf,
              const float* __restrict__ bias, void* __restrict__ outp, int N) {
    __shared__ _Float16 As[ROWS_PER_BLOCK * D_FEAT];     // 16 KB
    const int tid  = threadIdx.x;
    const int row0 = blockIdx.x * ROWS_PER_BLOCK;
    const bool full = (row0 + ROWS_PER_BLOCK) <= N;

    // --- stage A tile (rows row0 .. row0+63) into LDS as f16 ---
    // 8192 halves total; each thread handles 32 consecutive elements.
    {
        const int lbase = tid * 32;
        const int gbase = row0 * D_FEAT + lbase;
        _Float16* d = &As[lbase];
        if (full) {
#pragma unroll
            for (int j = 0; j < 4; ++j) {
                if constexpr (sizeof(SrcT) == 4) {
                    const float4* s = (const float4*)(A + gbase + j * 8);
                    float4 a0 = s[0], a1 = s[1];
                    _Float16* dj = d + j * 8;
                    dj[0] = (_Float16)a0.x; dj[1] = (_Float16)a0.y;
                    dj[2] = (_Float16)a0.z; dj[3] = (_Float16)a0.w;
                    dj[4] = (_Float16)a1.x; dj[5] = (_Float16)a1.y;
                    dj[6] = (_Float16)a1.z; dj[7] = (_Float16)a1.w;
                } else {
                    *(v8h*)(d + j * 8) = *(const v8h*)(A + gbase + j * 8);
                }
            }
        } else {
            const int lim = N * D_FEAT;
#pragma unroll
            for (int j = 0; j < 4; ++j) {
                _Float16* dj = d + j * 8;
                if (gbase + j * 8 + 7 < lim) {
                    if constexpr (sizeof(SrcT) == 4) {
                        const float4* s = (const float4*)(A + gbase + j * 8);
                        float4 a0 = s[0], a1 = s[1];
                        dj[0] = (_Float16)a0.x; dj[1] = (_Float16)a0.y;
                        dj[2] = (_Float16)a0.z; dj[3] = (_Float16)a0.w;
                        dj[4] = (_Float16)a1.x; dj[5] = (_Float16)a1.y;
                        dj[6] = (_Float16)a1.z; dj[7] = (_Float16)a1.w;
                    } else {
                        *(v8h*)dj = *(const v8h*)(A + gbase + j * 8);
                    }
                } else {
#pragma unroll
                    for (int i = 0; i < 8; ++i) dj[i] = (_Float16)0.0f;
                }
            }
        }
    }
    __syncthreads();

    const int lane = tid & 31;
    const int wv   = tid >> 5;               // N-tile 0..7
    const int rowA = lane & 15;
    const int kb   = (lane >> 4) << 3;       // 0 or 8

    v8f acc[4] = {};
#pragma unroll
    for (int kt = 0; kt < 4; ++kt) {
        const int kOff = kt * 32;
        // one B fragment per K-step, reused across 4 M-tiles
        v16h b = *(const v16h*)(wf + (((wv * 4 + kt) * 32 + lane) << 4));
        // load all 4 A fragments first (independent regs -> one ds clause)
        v16h a[4];
#pragma unroll
        for (int mt = 0; mt < 4; ++mt) {
            const int abase = (mt * 16 + rowA) * D_FEAT + kOff + kb;
            v8h lo = *(const v8h*)&As[abase];
            v8h hi = *(const v8h*)&As[abase + 16];
            a[mt] = __builtin_shufflevector(lo, hi,
                    0,1,2,3,4,5,6,7,8,9,10,11,12,13,14,15);
        }
        // 4 back-to-back WMMAs (distinct D, shared B -> no data hazard)
#pragma unroll
        for (int mt = 0; mt < 4; ++mt) {
            acc[mt] = __builtin_amdgcn_wmma_f32_16x16x32_f16(
                false, a[mt], false, b, (short)0, acc[mt], false, false);
        }
    }

    // --- epilogue: D layout (lane<16 -> M=r, lane>=16 -> M=r+8), col = lane&15
    const int   col = wv * 16 + (lane & 15);
    const float bv  = bias[col];
#pragma unroll
    for (int mt = 0; mt < 4; ++mt) {
        const int rbase = row0 + mt * 16 + ((lane >> 4) << 3);
        if (full) {
#pragma unroll
            for (int r = 0; r < 8; ++r) {
                float v = acc[mt][r] + bv;
                if (RELU) v = fmaxf(v, 0.0f);
                const int o = (rbase + r) * D_FEAT + col;
                if (OUT_F16) ((_Float16*)outp)[o] = (_Float16)v;
                else         ((float*)outp)[o]    = v;
            }
        } else {
#pragma unroll
            for (int r = 0; r < 8; ++r) {
                if (rbase + r < N) {
                    float v = acc[mt][r] + bv;
                    if (RELU) v = fmaxf(v, 0.0f);
                    const int o = (rbase + r) * D_FEAT + col;
                    if (OUT_F16) ((_Float16*)outp)[o] = (_Float16)v;
                    else         ((float*)outp)[o]    = v;
                }
            }
        }
    }
}

// ---------------------------------------------------------------------------
extern "C" void kernel_launch(void* const* d_in, const int* in_sizes, int n_in,
                              void* d_out, int out_size, void* d_ws, size_t ws_size,
                              hipStream_t stream) {
    const float*     x  = (const float*)d_in[0];
    const long long* ei = (const long long*)d_in[1];   // int64 edge_index [2, E]
    const float*     w1 = (const float*)d_in[2];
    const float*     b1 = (const float*)d_in[3];
    const float*     w2 = (const float*)d_in[4];
    const float*     b2 = (const float*)d_in[5];
    float*           out = (float*)d_out;

    const int N = in_sizes[0] / D_FEAT;   // 50000
    const int E = in_sizes[1] / 2;        // 600000

    // workspace layout (offsets 32B-aligned; N*768 divisible by 32 always)
    char* ws = (char*)d_ws;
    float*    h0  = (float*)ws;                                  // N*128 f32
    _Float16* h1  = (_Float16*)(ws + (size_t)N * D_FEAT * 4);    // N*128 f16
    _Float16* w1f = (_Float16*)(ws + (size_t)N * D_FEAT * 6);    // 128*128 f16
    _Float16* w2f = w1f + D_FEAT * D_FEAT;

    // 1) h0 = x
    {
        int n4 = N * D_FEAT / 4;
        init_h0<<<(n4 + 255) / 256, 256, 0, stream>>>((const float4*)x, (float4*)h0, n4);
    }
    // 2) weight fragment prep
    prep_w<<<4, 256, 0, stream>>>(w1, w1f);
    prep_w<<<4, 256, 0, stream>>>(w2, w2f);
    // 3) scatter-add over edges
    {
        long threads = (long)E * 32;
        scatter_edges<<<(int)((threads + 255) / 256), 256, 0, stream>>>(x, ei, h0, E);
    }
    // 4) h1 = relu(h0 @ w1 + b1), f16 out
    const int mblocks = (N + ROWS_PER_BLOCK - 1) / ROWS_PER_BLOCK;
    gemm_gin<float, true, true><<<mblocks, 256, 0, stream>>>(h0, w1f, b1, (void*)h1, N);
    // 5) out = h1 @ w2 + b2, f32 out
    gemm_gin<_Float16, false, false><<<mblocks, 256, 0, stream>>>(h1, w2f, b2, (void*)out, N);
}
